// GeometricAttention_8186207666909
// MI455X (gfx1250) — matrix operand order repaired and checked by reading, not compiled, not run on hardware
//
#include <hip/hip_runtime.h>
#include <hip/hip_bf16.h>

typedef __bf16 bf16;
typedef __attribute__((ext_vector_type(16))) __bf16 v16bf;
typedef __attribute__((ext_vector_type(8)))  __bf16 v8bf;
typedef __attribute__((ext_vector_type(8)))  float  v8f;
typedef __attribute__((ext_vector_type(4)))  unsigned int u32x4;
typedef __attribute__((ext_vector_type(8)))  int i32x8;
typedef __attribute__((ext_vector_type(4)))  int i32x4;

#if __has_include(<hip/amd_detail/amd_gfx1250_TDM.h>)
#define TDM_6ARG 1
#endif
#if __has_builtin(__builtin_amdgcn_tensor_load_to_lds)
#define HAVE_TDM 1
#endif

#define DEV __device__ __forceinline__

DEV v8f wmma_bf16(v16bf a, v16bf b, v8f c) {
  // D = A(16x32 bf16) * B(32x16 bf16) + C(16x16 f32)
  return __builtin_amdgcn_wmma_f32_16x16x32_bf16(
      /*neg_a=*/false, a, /*neg_b=*/false, b,
      /*c_mod=*/(short)0, c, /*reuse_a=*/false, /*reuse_b=*/false);
}

DEV v8f zero8() {
  v8f z;
#pragma unroll
  for (int i = 0; i < 8; ++i) z[i] = 0.0f;
  return z;
}

// A-matrix 16x32 bf16 fragment (row-major source, ld elements):
// lane m = lane&15 is the row; lane half selects K 0-7/16-23 vs 8-15/24-31.
DEV v16bf load_afrag(const bf16* A, int row, int lda, int k0, int khalf) {
  union { v16bf v; v8bf h[2]; } u;
  const bf16* p = A + (size_t)row * lda + k0 + khalf * 8;
  u.h[0] = *(const v8bf*)(p);        // elements 0..7  -> K = k0 + khalf*8 + e
  u.h[1] = *(const v8bf*)(p + 16);   // elements 8..15 -> K = k0 + 16 + khalf*8 + e
  return u.v;
}

// Same fragment pattern but from a 16x32 row-major LDS tile.
DEV v16bf load_afrag_lds(const bf16* base, int row, int khalf) {
  union { v16bf v; v8bf h[2]; } u;
  const bf16* p = base + row * 32 + khalf * 8;
  u.h[0] = *(const v8bf*)(p);
  u.h[1] = *(const v8bf*)(p + 16);
  return u.v;
}

// B-matrix 32x16 bf16 fragment from a transposed (N x K) source:
// lane n = lane&15 is the output column; elements 0..7 -> K = k0 + khalf*8 + e, 8..15 -> K = k0 + 16 + khalf*8 + e.
DEV v16bf load_bfrag(const bf16* Bt, int col, int ldb, int k0, int khalf) {
  union { v16bf v; v8bf h[2]; } u;
  const bf16* p = Bt + (size_t)col * ldb + k0 + khalf * 8;
  u.h[0] = *(const v8bf*)(p);
  u.h[1] = *(const v8bf*)(p + 16);
  return u.v;
}

#if HAVE_TDM
// TDM: DMA a 16-row x 32-col bf16 tile (row stride = K elements) into LDS.
// D# built per CDNA5 ISA 8.3/8.4: count=1, type=2(image), data_size=1(2B),
// tensor_dim0=K, tensor_dim1=Mtot, tile_dim0=32, tile_dim1=16, dim0_stride=K.
DEV void tdm_load_a_tile(const bf16* gsrc, unsigned lds_off, int K, int Mtot) {
  size_t ga = (size_t)gsrc;
  u32x4 g0;
  g0[0] = 1u;                                                   // count=1
  g0[1] = lds_off;                                              // lds_addr (bytes)
  g0[2] = (unsigned)ga;                                         // global_addr[31:0]
  g0[3] = (unsigned)((ga >> 32) & 0x01FFFFFFu) | 0x80000000u;   // addr[56:32] | type=2
  i32x8 g1;
  g1[0] = (int)(1u << 16);                                      // data_size = 2 bytes
  g1[1] = (int)(((unsigned)K & 0xFFFFu) << 16);                 // tensor_dim0[15:0] @ bit48
  g1[2] = (int)(((unsigned)K >> 16) | (((unsigned)Mtot & 0xFFFFu) << 16));
  g1[3] = (int)(((unsigned)Mtot >> 16) | (32u << 16));          // tile_dim0 = 32
  g1[4] = 16;                                                   // tile_dim1 = 16
  g1[5] = (int)(unsigned)K;                                     // tensor_dim0_stride lo32
  g1[6] = 0;
  g1[7] = 0;
  i32x4 z4 = {0, 0, 0, 0};
#if TDM_6ARG
  i32x8 z8 = {0, 0, 0, 0, 0, 0, 0, 0};
  __builtin_amdgcn_tensor_load_to_lds(g0, g1, z4, z4, z8, 0);
#else
  __builtin_amdgcn_tensor_load_to_lds(g0, g1, z4, z4, 0);
#endif
}
#endif

// ---------------------------------------------------------------------------
// Elementwise converters
// ---------------------------------------------------------------------------
__global__ __launch_bounds__(256) void cvt_f32_bf16(const float* in, bf16* out, int n) {
  int i = blockIdx.x * 256 + threadIdx.x;
  if (i < n) out[i] = (bf16)in[i];
}

// out[n*K + k] = (bf16) in[k*N + n]   (transpose K x N fp32 -> N x K bf16)
__global__ __launch_bounds__(256) void transp_f32_bf16(const float* in, bf16* out, int K, int N) {
  int i = blockIdx.x * 256 + threadIdx.x;
  if (i >= N * K) return;
  int k = i % K;
  int n = i / K;
  out[i] = (bf16)in[(size_t)k * N + n];
}

// ---------------------------------------------------------------------------
// GEMM: C[MxN] = A[MxK] (bf16 row-major) * Bt[NxK] (bf16, transposed weights)
// Block = 16 rows x 512 cols (8 waves x 64 cols). The 16x32 A tile per K-step
// is DMA'd into LDS by the Tensor Data Mover (double buffered), shared by all
// 8 waves; B fragments stream from L2 as contiguous b128 loads.
// ---------------------------------------------------------------------------
__global__ __launch_bounds__(256) void gemm_bf16_tn(const bf16* __restrict__ A,
                                                    const bf16* __restrict__ Bt,
                                                    float* __restrict__ C,
                                                    int M, int N, int K) {
  int lane  = threadIdx.x & 31;
  int wave  = threadIdx.x >> 5;
  int nloc  = lane & 15;
  int khalf = lane >> 4;
  int tilesN = N >> 9;                 // 512-wide block columns
  int bm = blockIdx.x / tilesN;
  int bn = blockIdx.x % tilesN;
  int colbase = bn * 512 + wave * 64;
  const bf16* Ablk = A + (size_t)bm * 16 * K;

  __shared__ bf16 atile[2][16 * 32];   // double-buffered 1 KB tiles at LDS 0 / 1024

  v8f acc[4];
#pragma unroll
  for (int s = 0; s < 4; ++s) acc[s] = zero8();

#if HAVE_TDM
  if (wave == 0) tdm_load_a_tile(Ablk, 0u, K, M);
#endif

  for (int k0 = 0; k0 < K; k0 += 32) {
    int p = (k0 >> 5) & 1;
    bool more = (k0 + 32) < K;
#if HAVE_TDM
    if (wave == 0) {
      if (more) {
        tdm_load_a_tile(Ablk + k0 + 32, (unsigned)((p ^ 1) * 1024), K, M);
        __builtin_amdgcn_s_wait_tensorcnt(1);   // previous tile complete
      } else {
        __builtin_amdgcn_s_wait_tensorcnt(0);
      }
    }
#else
    // Fallback: wave 0 stages the A tile with plain loads (row-half per lane).
    if (wave == 0) {
      const bf16* src = Ablk + (size_t)nloc * K + k0 + khalf * 16;
      *(v16bf*)(&atile[p][nloc * 32 + khalf * 16]) = *(const v16bf*)(src);
    }
    (void)more;
#endif
    __syncthreads();                   // tile p ready for all waves

    v16bf bfr[4];
#pragma unroll
    for (int s = 0; s < 4; ++s)
      bfr[s] = load_bfrag(Bt, colbase + s * 16 + nloc, K, k0, khalf);
    v16bf af = load_afrag_lds(&atile[p][0], nloc, khalf);
#pragma unroll
    for (int s = 0; s < 4; ++s) acc[s] = wmma_bf16(af, bfr[s], acc[s]);

    __syncthreads();                   // done reading tile p before it is reused
  }

#pragma unroll
  for (int s = 0; s < 4; ++s) {
#pragma unroll
    for (int i = 0; i < 8; ++i) {
      int row = bm * 16 + i + khalf * 8;
      int col = colbase + s * 16 + nloc;
      C[(size_t)row * N + col] = acc[s][i];
    }
  }
}

// ---------------------------------------------------------------------------
// RoPE + split + layout: qkv f32 [b][t][3][h][d] ->
//   qb,kb bf16 [b][h][t][d] (rope applied), vT bf16 [b][h][d][t]
// ---------------------------------------------------------------------------
__global__ __launch_bounds__(256) void rope_split(const float* __restrict__ qkv,
                                                  bf16* __restrict__ qb,
                                                  bf16* __restrict__ kb,
                                                  bf16* __restrict__ vT) {
  int idx = blockIdx.x * 256 + threadIdx.x;     // 2*2048*8*128 threads
  int d  = idx & 127;
  int nh = (idx >> 7) & 7;
  int t  = (idx >> 10) & 2047;
  int b  = idx >> 21;
  const float* base = qkv + ((size_t)(b * 2048 + t)) * 3072 + nh * 128;
  float qv = base[d];
  float kv = base[1024 + d];
  float vv = base[2048 + d];
  int j = d & 63;
  float inv = __powf(10000.0f, -(float)j * (1.0f / 64.0f));
  float ang = (float)t * inv;
  float c = __cosf(ang), s = __sinf(ang);
  float qpart, kpart;
  if (d < 64) { qpart = -base[d + 64];        kpart = -base[1024 + d + 64]; }
  else        { qpart =  base[d - 64];        kpart =  base[1024 + d - 64]; }
  float qr = qv * c + qpart * s;
  float kr = kv * c + kpart * s;
  size_t h = (size_t)(b * 8 + nh);
  qb[(h * 2048 + t) * 128 + d] = (bf16)qr;
  kb[(h * 2048 + t) * 128 + d] = (bf16)kr;
  vT[(h * 128 + d) * 2048 + t] = (bf16)vv;
}

// ---------------------------------------------------------------------------
// Flash attention (causal, online softmax). One wave per 16-query tile.
//   qb,kb: [b][h][T][D] bf16   vT: [b][h][D][T] bf16   aout f32: [b][t][h*D]
// ---------------------------------------------------------------------------
__global__ __launch_bounds__(32) void flash_attn(const bf16* __restrict__ qb,
                                                 const bf16* __restrict__ kb,
                                                 const bf16* __restrict__ vT,
                                                 float* __restrict__ aout) {
  const int T = 2048, D = 128;
  int tile = blockIdx.x & 127;
  int nh   = (blockIdx.x >> 7) & 7;
  int b    = blockIdx.x >> 10;
  int lane  = threadIdx.x & 31;
  int nloc  = lane & 15;
  int khalf = lane >> 4;
  const bf16* qh = qb + ((size_t)(b * 8 + nh)) * T * D;
  const bf16* kh = kb + ((size_t)(b * 8 + nh)) * T * D;
  const bf16* vh = vT + ((size_t)(b * 8 + nh)) * D * T;
  int q0 = tile * 16;

  v16bf qf[4];
#pragma unroll
  for (int kk = 0; kk < 4; ++kk) qf[kk] = load_afrag(qh, q0 + nloc, D, kk * 32, khalf);

  v8f oacc[8];
#pragma unroll
  for (int t = 0; t < 8; ++t) oacc[t] = zero8();
  float mst[8], lst[8];
#pragma unroll
  for (int i = 0; i < 8; ++i) { mst[i] = -3.0e38f; lst[i] = 0.0f; }

  __shared__ bf16 p_lds[16 * 48];   // 16 rows, stride 48 (16B-aligned sub-rows)
  const float scale = 0.08838834764831845f;   // 1/sqrt(128)

  int nblk = ((q0 + 15) >> 5) + 1;
  for (int blk = 0; blk < nblk; ++blk) {
    int kb0 = blk * 32;
    v8f sacc0 = zero8(), sacc1 = zero8();
#pragma unroll
    for (int kk = 0; kk < 4; ++kk) {
      v16bf b0 = load_bfrag(kh, kb0 + nloc,      D, kk * 32, khalf);
      v16bf b1 = load_bfrag(kh, kb0 + 16 + nloc, D, kk * 32, khalf);
      sacc0 = wmma_bf16(qf[kk], b0, sacc0);
      sacc1 = wmma_bf16(qf[kk], b1, sacc1);
    }
    // online softmax (row m = i + khalf*8 lives across 16 lanes of one half)
#pragma unroll
    for (int i = 0; i < 8; ++i) {
      int m  = i + khalf * 8;
      int qi = q0 + m;
      float s0 = (kb0 + nloc      <= qi) ? sacc0[i] * scale : -3.0e38f;
      float s1 = (kb0 + 16 + nloc <= qi) ? sacc1[i] * scale : -3.0e38f;
      float mx = fmaxf(s0, s1);
#pragma unroll
      for (int msk = 1; msk < 16; msk <<= 1) mx = fmaxf(mx, __shfl_xor(mx, msk, 32));
      float mnew = fmaxf(mst[i], mx);
      float p0 = __expf(s0 - mnew);
      float p1 = __expf(s1 - mnew);
      float rs = p0 + p1;
#pragma unroll
      for (int msk = 1; msk < 16; msk <<= 1) rs += __shfl_xor(rs, msk, 32);
      float corr = __expf(mst[i] - mnew);
      lst[i] = lst[i] * corr + rs;
      mst[i] = mnew;
#pragma unroll
      for (int t = 0; t < 8; ++t) oacc[t][i] *= corr;
      p_lds[m * 48 + nloc]      = (bf16)p0;
      p_lds[m * 48 + 16 + nloc] = (bf16)p1;
    }
    __syncthreads();
    // reshape P: C-layout -> A-matrix 16x32 fragment via LDS
    union { v16bf v; v8bf h[2]; } pu;
    pu.h[0] = *(const v8bf*)(&p_lds[nloc * 48 + khalf * 8]);
    pu.h[1] = *(const v8bf*)(&p_lds[nloc * 48 + 16 + khalf * 8]);
    v16bf pf = pu.v;
#pragma unroll
    for (int t = 0; t < 8; ++t) {
      v16bf vf = load_bfrag(vh, t * 16 + nloc, T, kb0, khalf);
      oacc[t] = wmma_bf16(pf, vf, oacc[t]);
    }
    __syncthreads();
  }

#pragma unroll
  for (int t = 0; t < 8; ++t) {
#pragma unroll
    for (int i = 0; i < 8; ++i) {
      int m = i + khalf * 8;
      float o = oacc[t][i] / lst[i];
      aout[((size_t)(b * T + q0 + m)) * 1024 + nh * 128 + t * 16 + nloc] = o;
    }
  }
}

// ---------------------------------------------------------------------------
// Geometric blade mixing epilogue: out = proj + mixed(proj, alpha)
// ---------------------------------------------------------------------------
__global__ __launch_bounds__(256) void blade_mix(const float* __restrict__ proj,
                                                 const float* __restrict__ alpha,
                                                 float* __restrict__ out) {
  int idx = blockIdx.x * 256 + threadIdx.x;  // 2*2048*128 threads
  int d  = idx & 127;
  int bt = idx >> 7;
  const float* p = proj + (size_t)bt * 1024 + d;
  float o[8];
#pragma unroll
  for (int h = 0; h < 8; ++h) o[h] = p[h * 128];
  const int idx2bm[8] = {0, 1, 2, 4, 3, 5, 6, 7};  // self-inverse permutation
  float mixed[8] = {0, 0, 0, 0, 0, 0, 0, 0};
#pragma unroll
  for (int i = 1; i < 8; ++i) {
#pragma unroll
    for (int j = 1; j < 8; ++j) {
      if (i == j) continue;
      int a = idx2bm[i], bb = idx2bm[j];
      int s = 0;
      for (int aa = a >> 1; aa; aa >>= 1) s += __popc(aa & bb);
      int k = idx2bm[a ^ bb];          // bm2idx == idx2bm
      float sign = (s & 1) ? -1.0f : 1.0f;
      mixed[k] += o[i] * o[j] * alpha[i * 8 + j] * sign;
    }
  }
  float* q = out + (size_t)bt * 1024 + d;
#pragma unroll
  for (int h = 0; h < 8; ++h) q[h * 128] = o[h] + mixed[h];
}

// ---------------------------------------------------------------------------
extern "C" void kernel_launch(void* const* d_in, const int* in_sizes, int n_in,
                              void* d_out, int out_size, void* d_ws, size_t ws_size,
                              hipStream_t stream) {
  (void)in_sizes; (void)n_in; (void)out_size; (void)ws_size;
  const float* mv    = (const float*)d_in[0];   // (2,2048,8,128) f32
  const float* Wqkv  = (const float*)d_in[1];   // (1024,3072) f32
  const float* Wo    = (const float*)d_in[2];   // (1024,1024) f32
  const float* alpha = (const float*)d_in[3];   // (8,8) f32
  float* out = (float*)d_out;

  char* ws = (char*)d_ws;
  size_t off = 0;
  auto take = [&](size_t bytes) -> void* {
    void* p = ws + off;
    off += (bytes + 255) & ~(size_t)255;
    return p;
  };
  bf16*  xb    = (bf16*) take(4096ull * 1024 * 2);
  bf16*  wqkvT = (bf16*) take(3072ull * 1024 * 2);
  bf16*  woT   = (bf16*) take(1024ull * 1024 * 2);
  float* qkvf  = (float*)take(4096ull * 3072 * 4);
  bf16*  qb    = (bf16*) take(2ull * 8 * 2048 * 128 * 2);
  bf16*  kb    = (bf16*) take(2ull * 8 * 2048 * 128 * 2);
  bf16*  vT    = (bf16*) take(2ull * 8 * 2048 * 128 * 2);
  float* aoutf = (float*)take(4096ull * 1024 * 4);
  bf16*  aoutb = (bf16*) take(4096ull * 1024 * 2);
  float* proj  = (float*)take(4096ull * 1024 * 4);

  // 1) bf16 conversions (+ transposed weights for contiguous B-fragments)
  cvt_f32_bf16<<<4194304 / 256, 256, 0, stream>>>(mv, xb, 4194304);
  transp_f32_bf16<<<3145728 / 256, 256, 0, stream>>>(Wqkv, wqkvT, 1024, 3072);
  transp_f32_bf16<<<1048576 / 256, 256, 0, stream>>>(Wo, woT, 1024, 1024);

  // 2) QKV projection: (4096x1024) @ (1024x3072); blocks of 16 x 512
  gemm_bf16_tn<<<256 * 6, 256, 0, stream>>>(xb, wqkvT, qkvf, 4096, 3072, 1024);

  // 3) RoPE + split + layout for attention
  rope_split<<<4194304 / 256, 256, 0, stream>>>(qkvf, qb, kb, vT);

  // 4) Causal flash attention, one wave per 16-query tile per (b, head)
  flash_attn<<<2048, 32, 0, stream>>>(qb, kb, vT, aoutf);

  // 5) Output projection: (4096x1024) @ (1024x1024)
  cvt_f32_bf16<<<4194304 / 256, 256, 0, stream>>>(aoutf, aoutb, 4194304);
  gemm_bf16_tn<<<256 * 2, 256, 0, stream>>>(aoutb, woT, proj, 4096, 1024, 1024);

  // 6) Geometric blade mixing epilogue
  blade_mix<<<524288 / 256, 256, 0, stream>>>(proj, alpha, out);
}
